// LIM_DFPN_73383811220012
// MI455X (gfx1250) — matrix operand order repaired and checked by reading, not compiled, not run on hardware
//
#include <hip/hip_runtime.h>

// ---------------------------------------------------------------------------
// Types for CDNA5 WMMA (wave32): v_wmma_f32_16x16x32_bf16
// ---------------------------------------------------------------------------
typedef __bf16 bf16_t;
typedef __attribute__((ext_vector_type(16))) __bf16 v16bf;
typedef __attribute__((ext_vector_type(8)))  __bf16 v8bf;
typedef __attribute__((ext_vector_type(8)))  float  v8f;

__device__ __forceinline__ float ld_as_float(const float*  p) { return *p; }
__device__ __forceinline__ float ld_as_float(const bf16_t* p) { return (float)*p; }

// ---------------------------------------------------------------------------
// Implicit-GEMM convolution with WMMA, LDS-staged B, double buffered.
//   Wt   : [Cout, K] bf16 row-major, K = Cin*KH*KW (OIHW flattened)
//   X    : [NB, Cin, IH, IW]  (float or bf16)
//   Y    : [NB, Cout, OH, OW] float, bias added
//
// Block = 256 threads = 8 waves. Block computes 128 (Cout) x 64 (pixels);
// wave w owns coTile = blockIdx.x*128 + w*16. The 32(K)x64(px) B tile per
// K-step is gathered cooperatively into LDS, stored directly in the WMMA
// B-fragment layout ([subtile][lane][16]), double-buffered so the gather for
// step n+1 overlaps the 4 WMMAs of step n. grid = (Cout/128, Npix/64).
//
// ISA 7.12.2 layouts:
//   A 16x32 bf16 : lane&15 = row M; elems 0..15 -> K = 16*(lane>=16)+i (this code's order, as B)
//   B 32x16 bf16 : lane&15 = column N, elems 0..15 -> K = 16*(lane>=16)+i (this code's order, not ISA table order)
//   C 16x16 f32  : vgpr r, lane l: M = r + 8*(l>=16), N = l&15
// ---------------------------------------------------------------------------
template<int KH, int KW, typename TIN>
__global__ __launch_bounds__(256)
void conv_wmma_kernel(const bf16_t* __restrict__ Wt,
                      const TIN*   __restrict__ X,
                      const float* __restrict__ bias,
                      float*       __restrict__ Y,
                      int Cin, int IH, int IW,
                      int stride, int pad,
                      int OH, int OW, int Cout)
{
    __shared__ __align__(16) bf16_t Bsm[2 * 2048];   // 2 x (4 subtiles x 32 lanes x 16)

    const int K    = Cin * KH * KW;
    const int tid  = threadIdx.x;
    const int lane = tid & 31;
    const int wave = tid >> 5;
    const int row  = lane & 15;
    const int half = lane >> 4;
    const int coTile  = (blockIdx.x << 7) + (wave << 4);
    const int pixBase = blockIdx.y << 6;

    // ---------- fill-side per-thread constants (one pixel, 8 consecutive K) --
    const int f_i0  = (tid & 1) << 3;            // element 0 or 8
    const int f_lp  = (tid >> 1) & 31;           // fragment lane slot
    const int f_t   = tid >> 6;                  // subtile 0..3
    const int f_kh  = (f_lp >> 4) << 4;          // K half offset 0/16
    const int f_lds = f_t * 512 + f_lp * 16 + f_i0;
    {
    }
    int fpix = pixBase + f_t * 16 + (f_lp & 15);
    int f_ox = fpix % OW;  int fq = fpix / OW;
    int f_oy = fq % OH;    int f_n = fq / OH;
    const int f_iy0 = f_oy * stride - pad;
    const int f_ix0 = f_ox * stride - pad;
    const TIN* f_xb = X + (size_t)f_n * Cin * IH * IW;

    auto fill = [&](int buf, int k0) {
        v8bf v;
#pragma unroll
        for (int j = 0; j < 8; ++j) {
            int k = k0 + f_kh + f_i0 + j;
            float val;
            if (KH == 1 && KW == 1) {
                // all 1x1 convs here are stride 1 / pad 0: always in range
                val = ld_as_float(&f_xb[((size_t)k * IH + f_iy0) * IW + f_ix0]);
            } else {
                int ci = k / (KH * KW);
                int r  = k - ci * (KH * KW);
                int ky = r / KW;
                int kx = r - ky * KW;
                int iy = f_iy0 + ky, ix = f_ix0 + kx;
                bool inb = ((unsigned)iy < (unsigned)IH) && ((unsigned)ix < (unsigned)IW);
                val = inb ? ld_as_float(&f_xb[((size_t)ci * IH + iy) * IW + ix]) : 0.0f;
            }
            v[j] = (__bf16)val;
        }
        *(v8bf*)(&Bsm[buf * 2048 + f_lds]) = v;   // one 16B ds_store, conflict-free
    };

    // ---------- read-side per-wave pixel decode (for epilogue) --------------
    int n_[4], oy_[4], ox_[4];
#pragma unroll
    for (int t = 0; t < 4; ++t) {
        int p = pixBase + t * 16 + row;
        ox_[t] = p % OW; int q2 = p / OW;
        oy_[t] = q2 % OH; n_[t] = q2 / OH;
    }

    v8f acc[4] = {};
    const bf16_t* wrow = Wt + (size_t)(coTile + row) * K + half * 16;

    fill(0, 0);
    __syncthreads();

    for (int k0 = 0; k0 < K; k0 += 32) {
        const int cur = (k0 >> 5) & 1;
        if (k0 + 32 < K) fill(cur ^ 1, k0 + 32);   // overlap gather with WMMA

        __builtin_prefetch(wrow + k0 + 256, 0, 0);
        // ---- A fragment: contiguous bf16, two aligned 16B global loads
        v8bf alo = *(const v8bf*)(wrow + k0);
        v8bf ahi = *(const v8bf*)(wrow + k0 + 8);
        v16bf a;
#pragma unroll
        for (int i = 0; i < 8; ++i) { a[i] = alo[i]; a[i + 8] = ahi[i]; }

#pragma unroll
        for (int t = 0; t < 4; ++t) {
            // ---- B fragment: two ds_load_b128 from pre-swizzled LDS tile
            const bf16_t* fpb = &Bsm[cur * 2048 + t * 512 + lane * 16];
            v8bf blo = *(const v8bf*)fpb;
            v8bf bhi = *(const v8bf*)(fpb + 8);
            v16bf bb;
#pragma unroll
            for (int i = 0; i < 8; ++i) { bb[i] = blo[i]; bb[i + 8] = bhi[i]; }
            acc[t] = __builtin_amdgcn_wmma_f32_16x16x32_bf16(
                false, a, false, bb, (short)0, acc[t], false, false);
        }
        __syncthreads();
    }

    // ---- epilogue: bias + store (C layout per ISA)
#pragma unroll
    for (int t = 0; t < 4; ++t) {
        size_t base = (((size_t)n_[t] * Cout) * OH + oy_[t]) * OW + ox_[t];
#pragma unroll
        for (int r = 0; r < 8; ++r) {
            int co = coTile + half * 8 + r;
            Y[base + (size_t)co * OH * OW] = acc[t][r] + bias[co];
        }
    }
}

// ---------------------------------------------------------------------------
// Boundary aggregation input builder: writes concat([x, cummaxH, cummaxH_rev,
// cummaxW, cummaxW_rev], axis=1) as bf16, layout [n][5C][S][S].
// One block per (n,c) plane; thread t scans row t and column t.
// ---------------------------------------------------------------------------
__global__ void boundary_concat_kernel(const float* __restrict__ X,
                                       bf16_t* __restrict__ Y,
                                       int C, int S)
{
    const size_t SS = (size_t)S * S;
    const int nc = blockIdx.x;
    const int n  = nc / C;
    const int c  = nc % C;
    const float* p = X + (size_t)nc * SS;
    bf16_t* o = Y + ((size_t)n * 5 * C + c) * SS;   // block j lives at +j*C*SS
    const size_t blk = (size_t)C * SS;
    const int t = threadIdx.x;
    if (t >= S) return;

    // block 0: copy x (row t)
    for (int x = 0; x < S; ++x) o[(size_t)t * S + x] = (__bf16)p[t * S + x];
    // block 1: prefix max along H (column t)
    float m = -3.402823466e38f;
    for (int y = 0; y < S; ++y) { m = fmaxf(m, p[y * S + t]); o[blk + (size_t)y * S + t] = (__bf16)m; }
    // block 2: suffix max along H (column t)
    m = -3.402823466e38f;
    for (int y = S - 1; y >= 0; --y) { m = fmaxf(m, p[y * S + t]); o[2 * blk + (size_t)y * S + t] = (__bf16)m; }
    // block 3: prefix max along W (row t)
    m = -3.402823466e38f;
    for (int x = 0; x < S; ++x) { m = fmaxf(m, p[t * S + x]); o[3 * blk + (size_t)t * S + x] = (__bf16)m; }
    // block 4: suffix max along W (row t)
    m = -3.402823466e38f;
    for (int x = S - 1; x >= 0; --x) { m = fmaxf(m, p[t * S + x]); o[4 * blk + (size_t)t * S + x] = (__bf16)m; }
}

// ---------------------------------------------------------------------------
// Elementwise helpers
// ---------------------------------------------------------------------------
__global__ void cvt_f32_bf16_kernel(const float* __restrict__ src,
                                    bf16_t* __restrict__ dst, int n)
{
    int i = blockIdx.x * 256 + threadIdx.x;
    if (i < n) dst[i] = (__bf16)src[i];
}

// dst[n,c,y,x] += src[n,c,y>>shift,x>>shift]   (nearest-neighbor upsample add)
__global__ void add_up_kernel(float* __restrict__ dst, const float* __restrict__ src,
                              int C, int OH, int OW, int shift, int total)
{
    int i = blockIdx.x * 256 + threadIdx.x;
    if (i >= total) return;
    int x = i % OW; int q = i / OW;
    int y = q % OH; q /= OH;
    int c = q % C;  int n = q / C;
    int sh = OH >> shift, sw = OW >> shift;
    dst[i] += src[(((size_t)n * C + c) * sh + (y >> shift)) * sw + (x >> shift)];
}

__global__ void add_inplace_kernel(float* __restrict__ dst,
                                   const float* __restrict__ src, int n)
{
    int i = blockIdx.x * 256 + threadIdx.x;
    if (i < n) dst[i] += src[i];
}

__global__ void relu_kernel(float* __restrict__ dst, const float* __restrict__ src, int n)
{
    int i = blockIdx.x * 256 + threadIdx.x;
    if (i < n) dst[i] = fmaxf(src[i], 0.0f);
}

// dst = g*pd + (1-g)*p
__global__ void merge_kernel(float* __restrict__ dst, const float* __restrict__ pd,
                             const float* __restrict__ p, const float* __restrict__ gamma,
                             int n)
{
    int i = blockIdx.x * 256 + threadIdx.x;
    if (i >= n) return;
    float g = gamma[0];
    dst[i] = g * pd[i] + (1.0f - g) * p[i];
}

// ---------------------------------------------------------------------------
// Host-side orchestration
// ---------------------------------------------------------------------------
extern "C" void kernel_launch(void* const* d_in, const int* in_sizes, int n_in,
                              void* d_out, int out_size, void* d_ws, size_t ws_size,
                              hipStream_t stream)
{
    (void)in_sizes; (void)n_in; (void)out_size; (void)ws_size;

    const float* C3 = (const float*)d_in[0];   // [8,256,64,64]
    const float* C4 = (const float*)d_in[1];   // [8,512,32,32]
    const float* C5 = (const float*)d_in[2];   // [8,1024,16,16]

    char* wsb = (char*)d_ws;
    size_t off = 0;
    auto alloc = [&](size_t bytes) -> void* {
        off = (off + 255) & ~(size_t)255;
        void* p = (void*)(wsb + off);
        off += bytes;
        return p;
    };
    auto ew = [](int n) { return dim3((unsigned)((n + 255) / 256)); };

    // ---- convert all conv weights to bf16 [Cout, K] ----
    // order: ba3 ba4 ba5 p5_1 p5_1d p5_2 p4_1 p4_1d p4_2 p3_1 p3_1d p3_2 p6 p7 c34 c45 c35
    const int    widx[17] = {3, 5, 7, 9, 11, 13, 15, 17, 19, 21, 23, 25, 27, 29, 31, 33, 35};
    const size_t wcnt[17] = {327680ull, 1310720ull, 5242880ull, 262144ull, 262144ull, 589824ull,
                             131072ull, 131072ull, 589824ull, 65536ull, 65536ull, 589824ull,
                             2359296ull, 589824ull, 589824ull, 262144ull, 262144ull};
    bf16_t* wb[17];
    for (int i = 0; i < 17; ++i) {
        wb[i] = (bf16_t*)alloc(wcnt[i] * sizeof(bf16_t));
        int n = (int)wcnt[i];
        cvt_f32_bf16_kernel<<<ew(n), 256, 0, stream>>>((const float*)d_in[widx[i]], wb[i], n);
    }
    enum { W_BA3 = 0, W_BA4, W_BA5, W_P51, W_P51D, W_P52, W_P41, W_P41D, W_P42,
           W_P31, W_P31D, W_P32, W_P6, W_P7, W_C34, W_C45, W_C35 };

    // ---- workspace buffers ----
    bf16_t* BAIN = (bf16_t*)alloc(41943040ull * sizeof(bf16_t)); // max of 3 levels, reused
    float* C3BA  = (float*)alloc(8388608ull * 4);
    float* C4BA  = (float*)alloc(4194304ull * 4);
    float* C5BA  = (float*)alloc(2097152ull * 4);
    float* P5pre = (float*)alloc(524288ull * 4);
    float* P5    = (float*)alloc(524288ull * 4);
    float* P4S   = (float*)alloc(2097152ull * 4);
    float* P4    = (float*)alloc(2097152ull * 4);
    float* P3S   = (float*)alloc(8388608ull * 4);
    float* P3    = (float*)alloc(8388608ull * 4);
    float* P3D   = (float*)alloc(8388608ull * 4);
    float* P3D1  = (float*)alloc(2097152ull * 4);
    float* P3D2  = (float*)alloc(524288ull * 4);
    float* P4D   = (float*)alloc(2097152ull * 4);
    float* P4D1  = (float*)alloc(524288ull * 4);
    float* P5D   = (float*)alloc(524288ull * 4);
    float* P6R   = (float*)alloc(131072ull * 4);

    // ---- boundary aggregation (sequential levels reuse BAIN) ----
    boundary_concat_kernel<<<dim3(8 * 256), 64, 0, stream>>>(C3, BAIN, 256, 64);
    conv_wmma_kernel<1, 1, bf16_t><<<dim3(2, 512), 256, 0, stream>>>(
        wb[W_BA3], BAIN, (const float*)d_in[4], C3BA, 1280, 64, 64, 1, 0, 64, 64, 256);

    boundary_concat_kernel<<<dim3(8 * 512), 32, 0, stream>>>(C4, BAIN, 512, 32);
    conv_wmma_kernel<1, 1, bf16_t><<<dim3(4, 128), 256, 0, stream>>>(
        wb[W_BA4], BAIN, (const float*)d_in[6], C4BA, 2560, 32, 32, 1, 0, 32, 32, 512);

    boundary_concat_kernel<<<dim3(8 * 1024), 16, 0, stream>>>(C5, BAIN, 1024, 16);
    conv_wmma_kernel<1, 1, bf16_t><<<dim3(8, 32), 256, 0, stream>>>(
        wb[W_BA5], BAIN, (const float*)d_in[8], C5BA, 5120, 16, 16, 1, 0, 16, 16, 1024);

    // ---- top-down FPN path ----
    conv_wmma_kernel<1, 1, float><<<dim3(2, 32), 256, 0, stream>>>(
        wb[W_P51], C5, (const float*)d_in[10], P5pre, 1024, 16, 16, 1, 0, 16, 16, 256);
    conv_wmma_kernel<3, 3, float><<<dim3(2, 32), 256, 0, stream>>>(
        wb[W_P52], P5pre, (const float*)d_in[14], P5, 256, 16, 16, 1, 1, 16, 16, 256);

    conv_wmma_kernel<1, 1, float><<<dim3(2, 128), 256, 0, stream>>>(
        wb[W_P41], C4, (const float*)d_in[16], P4S, 512, 32, 32, 1, 0, 32, 32, 256);
    add_up_kernel<<<ew(2097152), 256, 0, stream>>>(P4S, P5pre, 256, 32, 32, 1, 2097152);
    conv_wmma_kernel<3, 3, float><<<dim3(2, 128), 256, 0, stream>>>(
        wb[W_P42], P4S, (const float*)d_in[20], P4, 256, 32, 32, 1, 1, 32, 32, 256);

    conv_wmma_kernel<1, 1, float><<<dim3(2, 512), 256, 0, stream>>>(
        wb[W_P31], C3, (const float*)d_in[22], P3S, 256, 64, 64, 1, 0, 64, 64, 256);
    add_up_kernel<<<ew(8388608), 256, 0, stream>>>(P3S, P4S, 256, 64, 64, 1, 8388608);
    add_up_kernel<<<ew(8388608), 256, 0, stream>>>(P3S, P5pre, 256, 64, 64, 2, 8388608);
    conv_wmma_kernel<3, 3, float><<<dim3(2, 512), 256, 0, stream>>>(
        wb[W_P32], P3S, (const float*)d_in[26], P3, 256, 64, 64, 1, 1, 64, 64, 256);

    // ---- bottom-up dual path on BA features ----
    conv_wmma_kernel<1, 1, float><<<dim3(2, 512), 256, 0, stream>>>(
        wb[W_P31D], C3BA, (const float*)d_in[24], P3D, 256, 64, 64, 1, 0, 64, 64, 256);
    conv_wmma_kernel<3, 3, float><<<dim3(2, 128), 256, 0, stream>>>(
        wb[W_C34], P3D, (const float*)d_in[32], P3D1, 256, 64, 64, 2, 1, 32, 32, 256);
    conv_wmma_kernel<2, 2, float><<<dim3(2, 32), 256, 0, stream>>>(
        wb[W_C35], P3D1, (const float*)d_in[36], P3D2, 256, 32, 32, 2, 0, 16, 16, 256);

    conv_wmma_kernel<1, 1, float><<<dim3(2, 128), 256, 0, stream>>>(
        wb[W_P41D], C4BA, (const float*)d_in[18], P4D, 512, 32, 32, 1, 0, 32, 32, 256);
    add_inplace_kernel<<<ew(2097152), 256, 0, stream>>>(P4D, P3D1, 2097152);
    conv_wmma_kernel<2, 2, float><<<dim3(2, 32), 256, 0, stream>>>(
        wb[W_C45], P4D, (const float*)d_in[34], P4D1, 256, 32, 32, 2, 0, 16, 16, 256);

    conv_wmma_kernel<1, 1, float><<<dim3(2, 32), 256, 0, stream>>>(
        wb[W_P51D], C5BA, (const float*)d_in[12], P5D, 1024, 16, 16, 1, 0, 16, 16, 256);
    add_inplace_kernel<<<ew(524288), 256, 0, stream>>>(P5D, P4D1, 524288);
    add_inplace_kernel<<<ew(524288), 256, 0, stream>>>(P5D, P3D2, 524288);

    // ---- extra levels (direct to d_out) ----
    float* out   = (float*)d_out;
    float* O3    = out;
    float* O4    = out + 8388608;
    float* O5    = out + 10485760;
    float* P6out = out + 11010048;
    float* P7out = out + 11141120;

    conv_wmma_kernel<3, 3, float><<<dim3(2, 8), 256, 0, stream>>>(
        wb[W_P6], C5, (const float*)d_in[28], P6out, 1024, 16, 16, 2, 1, 8, 8, 256);
    relu_kernel<<<ew(131072), 256, 0, stream>>>(P6R, P6out, 131072);
    conv_wmma_kernel<3, 3, float><<<dim3(2, 2), 256, 0, stream>>>(
        wb[W_P7], P6R, (const float*)d_in[30], P7out, 256, 8, 8, 2, 1, 4, 4, 256);

    // ---- gated merges ----
    merge_kernel<<<ew(8388608), 256, 0, stream>>>(O3, P3D, P3, (const float*)d_in[37], 8388608);
    merge_kernel<<<ew(2097152), 256, 0, stream>>>(O4, P4D, P4, (const float*)d_in[38], 2097152);
    merge_kernel<<<ew(524288), 256, 0, stream>>>(O5, P5D, P5, (const float*)d_in[39], 524288);
}
